// TemporalLSTM_23184233464641
// MI455X (gfx1250) — matrix operand order repaired and checked by reading, not compiled, not run on hardware
//
#include <hip/hip_runtime.h>
#include <hip/hip_bf16.h>

// ---------------------------------------------------------------------------
// Problem constants (match reference)
// ---------------------------------------------------------------------------
#define BATCH   16384
#define HIDDEN  4096
#define CHUNKS  4
#define CHUNK   1024      // HIDDEN / CHUNKS
#define LH      256       // lstm hidden
#define G4      1024      // 4*LH (gate width)
#define TOTAL   2048      // CHUNKS * 2 * LH

typedef __attribute__((ext_vector_type(16))) __bf16 v16bf;
typedef __attribute__((ext_vector_type(8)))  __bf16 v8bf;
typedef __attribute__((ext_vector_type(8)))  float  v8f;

// ---------------------------------------------------------------------------
// WMMA helpers (CDNA5 bf16 16x16x32, fp32 accumulate)
// ---------------------------------------------------------------------------
__device__ __forceinline__ v8f vzero8() {
    v8f z;
#pragma unroll
    for (int i = 0; i < 8; ++i) z[i] = 0.0f;
    return z;
}

__device__ __forceinline__ v8f wmma_bf16(v16bf a, v16bf b, v8f c) {
    return __builtin_amdgcn_wmma_f32_16x16x32_bf16(
        /*neg_a=*/false, a, /*neg_b=*/false, b,
        /*c_mod=*/(short)0, c, /*reuse_a=*/false, /*reuse_b=*/false);
}

// A-operand fragment: 16x32 bf16 tile, lane l holds row (l%16).
// Elements 0..7  = K[kbase + 8*half + 0..7]
// Elements 8..15 = K[kbase + 16 + 8*half + 0..7]   (half = l/16)
__device__ __forceinline__ v16bf load_fragA(const __bf16* base, int ld,
                                            int row, int kbase, int half) {
    const __bf16* p = base + (size_t)row * ld + kbase + half * 8;
    v8bf lo = *reinterpret_cast<const v8bf*>(p);
    v8bf hi = *reinterpret_cast<const v8bf*>(p + 16);
    v16bf r;
#pragma unroll
    for (int i = 0; i < 8; ++i) { r[i] = lo[i]; r[i + 8] = hi[i]; }
    return r;
}

// B-operand fragment: 32x16 bf16 tile; we feed B^T rows (row-major in K).
// Lane l holds column (l%16); elements 0..7 = K[kbase + 8*half + 0..7], 8..15 = K[kbase + 16 + 8*half + 0..7].
__device__ __forceinline__ v16bf load_fragB(const __bf16* base, int ld,
                                            int row, int kbase, int half) {
    const __bf16* p = base + (size_t)row * ld + kbase + half * 8;
    v8bf lo = *reinterpret_cast<const v8bf*>(p);
    v8bf hi = *reinterpret_cast<const v8bf*>(p + 16);
    v16bf r;
#pragma unroll
    for (int i = 0; i < 8; ++i) { r[i] = lo[i]; r[i + 8] = hi[i]; }
    return r;
}

// C/D tile store: vgpr r holds M = r + 8*half, lane holds N = l%16.
__device__ __forceinline__ void store_tile(float* C, int ldc, const float* bias,
                                           int mrow, int ncol, int l16, int half,
                                           v8f acc) {
    int col = ncol + l16;
    float bv = bias ? bias[col] : 0.0f;
#pragma unroll
    for (int r = 0; r < 8; ++r) {
        int row = mrow + r + 8 * half;
        C[(size_t)row * ldc + col] = acc[r] + bv;
    }
}

// ---------------------------------------------------------------------------
// CDNA5 async copy (global -> LDS, ASYNCcnt-tracked), per-lane 16 bytes.
// ---------------------------------------------------------------------------
__device__ __forceinline__ void async_copy_b128(void* lds_dst, const void* gsrc) {
    unsigned lds_off = (unsigned)(size_t)lds_dst;   // flat[31:0] == LDS offset
    asm volatile("global_load_async_to_lds_b128 %0, %1, off"
                 :: "v"(lds_off), "v"(gsrc) : "memory");
}
__device__ __forceinline__ void wait_async_le1() {
    asm volatile("s_wait_asynccnt 0x1" ::: "memory");
}
__device__ __forceinline__ void wait_async_le0() {
    asm volatile("s_wait_asynccnt 0x0" ::: "memory");
}

// ---------------------------------------------------------------------------
// GEMM: C[M,N] (fp32) = A[M,K](bf16) * B[N,K]^T (bf16) + bias[N]
// grid.x = M/32, grid.y = N/512; block = 256 threads (8 waves).
// Block shares a 32-row A tile staged in LDS by async copies (double-
// buffered, 32x32 bf16 per K-step); each wave computes a 32x64 output tile
// (8 WMMA accumulators) with B fragments streamed from global (L2-resident).
// ---------------------------------------------------------------------------
__global__ __launch_bounds__(256)
void gemm_bt_bf16(const __bf16* __restrict__ A, int lda,
                  const __bf16* __restrict__ Bm, int ldb,
                  const float* __restrict__ bias,
                  float* __restrict__ C, int ldc, int K) {
    __shared__ __align__(16) __bf16 As[2][32 * 32];   // double-buffered A tile

    const int tid  = threadIdx.x;
    const int lane = tid & 31;
    const int wid  = tid >> 5;
    const int half = lane >> 4;
    const int l16  = lane & 15;
    const int m0 = blockIdx.x * 32;
    const int n0 = (blockIdx.y * 8 + wid) * 64;

    // async fill lanes: tid<128 -> row r (0..31), quarter q (0..3), 16B each
    const int r = (tid >> 2) & 31;
    const int q = tid & 3;
    const __bf16* grow = A + (size_t)(m0 + r) * lda + q * 8;

    // prologue: fill buffer 0 with K-block 0
    if (tid < 128) async_copy_b128(&As[0][r * 32 + q * 8], grow);

    v8f acc[2][4];
#pragma unroll
    for (int i = 0; i < 2; ++i)
#pragma unroll
        for (int j = 0; j < 4; ++j) acc[i][j] = vzero8();

    for (int kk = 0; kk < K; kk += 32) {
        const int cur = (kk >> 5) & 1;
        const bool more = (kk + 32) < K;
        if (tid < 128) {
            if (more) {
                async_copy_b128(&As[cur ^ 1][r * 32 + q * 8], grow + kk + 32);
                wait_async_le1();          // fill(kk) done; fill(kk+32) in flight
            } else {
                wait_async_le0();
            }
        }
        __syncthreads();                   // A tile visible to all 8 waves

        v16bf a0 = load_fragA(As[cur], 32, l16,      0, half);
        v16bf a1 = load_fragA(As[cur], 32, 16 + l16, 0, half);
#pragma unroll
        for (int j = 0; j < 4; ++j) {
            v16bf b = load_fragB(Bm, ldb, n0 + j * 16 + l16, kk, half);
            acc[0][j] = wmma_bf16(a0, b, acc[0][j]);
            acc[1][j] = wmma_bf16(a1, b, acc[1][j]);
        }
        __syncthreads();                   // done reading before next overwrite
    }

#pragma unroll
    for (int j = 0; j < 4; ++j) {
        store_tile(C, ldc, bias, m0,      n0 + j * 16, l16, half, acc[0][j]);
        store_tile(C, ldc, bias, m0 + 16, n0 + j * 16, l16, half, acc[1][j]);
    }
}

// ---------------------------------------------------------------------------
// Bidirectional LSTM recurrence. One 1024-thread block owns 32 batch rows.
// h (bf16) and c (fp32) live in LDS; gates staged in LDS between the WMMA
// h@Whh^T GEMM and the elementwise cell update. No cross-block deps.
// Dynamic LDS: h 32*256*2 + c 32*256*4 + gates 32*1024*4 = 180224 B.
// ---------------------------------------------------------------------------
__global__ __launch_bounds__(1024)
void lstm_recurrence(const __bf16* __restrict__ WhhF,
                     const __bf16* __restrict__ WhhB,
                     const float* __restrict__ Gf,
                     const float* __restrict__ Gb,
                     __bf16* __restrict__ flat) {
    extern __shared__ char smem[];
    __bf16* hS = reinterpret_cast<__bf16*>(smem);                        // [32][256]
    float*  cS = reinterpret_cast<float*>(smem + 32 * 256 * 2);          // [32][256]
    float*  gS = reinterpret_cast<float*>(smem + 32 * 256 * 2 + 32 * 256 * 4); // [32][1024]

    const int tid  = threadIdx.x;
    const int lane = tid & 31;
    const int wid  = tid >> 5;           // 0..31 -> 32-col slice of G4
    const int half = lane >> 4;
    const int l16  = lane & 15;
    const int mbase = blockIdx.x * 32;

    for (int dir = 0; dir < 2; ++dir) {
        // reset state
        for (int i = tid; i < 32 * 256; i += 1024) {
            hS[i] = (__bf16)0.0f;
            cS[i] = 0.0f;
        }
        __syncthreads();

        const __bf16* Whh = dir ? WhhB : WhhF;
        const float*  G   = dir ? Gb   : Gf;

        for (int step = 0; step < CHUNKS; ++step) {
            const int t = dir ? (CHUNKS - 1 - step) : step;

            // gates(32 x 1024) = h(32 x 256) @ Whh^T  via WMMA
            const int n0 = wid * 32;
            v8f acc00 = vzero8(), acc01 = vzero8(), acc10 = vzero8(), acc11 = vzero8();
#pragma unroll
            for (int kk = 0; kk < LH; kk += 32) {
                v16bf a0 = load_fragA(hS, LH, l16,      kk, half);
                v16bf a1 = load_fragA(hS, LH, 16 + l16, kk, half);
                v16bf b0 = load_fragB(Whh, LH, n0 + l16,      kk, half);
                v16bf b1 = load_fragB(Whh, LH, n0 + 16 + l16, kk, half);
                acc00 = wmma_bf16(a0, b0, acc00);
                acc01 = wmma_bf16(a0, b1, acc01);
                acc10 = wmma_bf16(a1, b0, acc10);
                acc11 = wmma_bf16(a1, b1, acc11);
            }

            // add precomputed input-gate preactivations, stage gates in LDS
            const float* Gt = G + ((size_t)t * BATCH + mbase) * G4;
            {
                int col0 = n0 + l16, col1 = n0 + 16 + l16;
#pragma unroll
                for (int rr = 0; rr < 8; ++rr) {
                    int row0 = rr + 8 * half, row1 = 16 + rr + 8 * half;
                    gS[row0 * G4 + col0] = acc00[rr] + Gt[(size_t)row0 * G4 + col0];
                    gS[row0 * G4 + col1] = acc01[rr] + Gt[(size_t)row0 * G4 + col1];
                    gS[row1 * G4 + col0] = acc10[rr] + Gt[(size_t)row1 * G4 + col0];
                    gS[row1 * G4 + col1] = acc11[rr] + Gt[(size_t)row1 * G4 + col1];
                }
            }
            __syncthreads();

            // cell update: gates order i,f,g,o
            for (int u = tid; u < 32 * 256; u += 1024) {
                int m = u >> 8;
                int n = u & 255;
                float iv = gS[m * G4 + n];
                float fv = gS[m * G4 + 256 + n];
                float gv = gS[m * G4 + 512 + n];
                float ov = gS[m * G4 + 768 + n];
                float si = 1.0f / (1.0f + expf(-iv));
                float sf = 1.0f / (1.0f + expf(-fv));
                float so = 1.0f / (1.0f + expf(-ov));
                float c  = sf * cS[u] + si * tanhf(gv);
                float h  = so * tanhf(c);
                cS[u] = c;
                hS[u] = (__bf16)h;
                // flat[b][t*512 + dir*256 + n]
                flat[(size_t)(mbase + m) * TOTAL + t * 512 + dir * 256 + n] = (__bf16)h;
            }
            __syncthreads();
        }
    }
}

// ---------------------------------------------------------------------------
// LayerNorm + exact GELU + residual. One 256-thread block per row.
// ---------------------------------------------------------------------------
__global__ __launch_bounds__(256)
void ln_gelu_residual(const float* __restrict__ y, const float* __restrict__ x,
                      const float* __restrict__ g, const float* __restrict__ b,
                      float* __restrict__ out) {
    const int row = blockIdx.x;
    const int tid = threadIdx.x;
    const float* yr = y + (size_t)row * HIDDEN;

    float s = 0.0f, s2 = 0.0f;
    for (int i = tid; i < HIDDEN; i += 256) {
        float v = yr[i];
        s += v;
        s2 += v * v;
    }
    __shared__ float rs[256], rs2[256];
    rs[tid] = s; rs2[tid] = s2;
    __syncthreads();
    for (int o = 128; o > 0; o >>= 1) {
        if (tid < o) { rs[tid] += rs[tid + o]; rs2[tid] += rs2[tid + o]; }
        __syncthreads();
    }
    const float mean = rs[0] * (1.0f / HIDDEN);
    const float var  = rs2[0] * (1.0f / HIDDEN) - mean * mean;
    const float inv  = rsqrtf(var + 1e-5f);

    for (int i = tid; i < HIDDEN; i += 256) {
        float v  = (yr[i] - mean) * inv * g[i] + b[i];
        float ge = 0.5f * v * (1.0f + erff(v * 0.70710678118654752f));
        out[(size_t)row * HIDDEN + i] = x[(size_t)row * HIDDEN + i] + ge;
    }
}

// ---------------------------------------------------------------------------
// Small utility kernels
// ---------------------------------------------------------------------------
__global__ void f2bf_kernel(const float* __restrict__ in, __bf16* __restrict__ out,
                            long n) {
    long i = (long)blockIdx.x * blockDim.x + threadIdx.x;
    if (i < n) out[i] = (__bf16)in[i];
}

__global__ void addvec_kernel(const float* __restrict__ a, const float* __restrict__ b,
                              float* __restrict__ o, int n) {
    int i = blockIdx.x * blockDim.x + threadIdx.x;
    if (i < n) o[i] = a[i] + b[i];
}

// ---------------------------------------------------------------------------
// Host launch
// ---------------------------------------------------------------------------
extern "C" void kernel_launch(void* const* d_in, const int* in_sizes, int n_in,
                              void* d_out, int out_size, void* d_ws, size_t ws_size,
                              hipStream_t stream) {
    const float* x     = (const float*)d_in[0];
    const float* Wih_f = (const float*)d_in[1];
    const float* Whh_f = (const float*)d_in[2];
    const float* bih_f = (const float*)d_in[3];
    const float* bhh_f = (const float*)d_in[4];
    const float* Wih_b = (const float*)d_in[5];
    const float* Whh_b = (const float*)d_in[6];
    const float* bih_b = (const float*)d_in[7];
    const float* bhh_b = (const float*)d_in[8];
    const float* projW = (const float*)d_in[9];
    const float* projb = (const float*)d_in[10];
    const float* ln_g  = (const float*)d_in[11];
    const float* ln_b  = (const float*)d_in[12];
    float* out = (float*)d_out;

    // workspace layout (bytes, 256-aligned sizes)
    char* ws = (char*)d_ws;
    constexpr size_t SZ_XB    = (size_t)BATCH * HIDDEN * 2;   // 134217728
    constexpr size_t SZ_WIH   = (size_t)G4 * CHUNK * 2;       // 2097152
    constexpr size_t SZ_WHH   = (size_t)G4 * LH * 2;          // 524288
    constexpr size_t SZ_PROJ  = (size_t)HIDDEN * TOTAL * 2;   // 16777216
    constexpr size_t SZ_BSUM  = (size_t)G4 * 4;               // 4096
    constexpr size_t SZ_G     = (size_t)CHUNKS * BATCH * G4 * 4; // 268435456
    constexpr size_t SZ_FLAT  = (size_t)BATCH * TOTAL * 2;    // 67108864

    size_t off = 0;
    __bf16* xb     = (__bf16*)(ws + off); off += SZ_XB;
    __bf16* WihFb  = (__bf16*)(ws + off); off += SZ_WIH;
    __bf16* WihBb  = (__bf16*)(ws + off); off += SZ_WIH;
    __bf16* WhhFb  = (__bf16*)(ws + off); off += SZ_WHH;
    __bf16* WhhBb  = (__bf16*)(ws + off); off += SZ_WHH;
    __bf16* projWb = (__bf16*)(ws + off); off += SZ_PROJ;
    float*  bsumF  = (float*)(ws + off);  off += SZ_BSUM;
    float*  bsumB  = (float*)(ws + off);  off += SZ_BSUM;
    float*  Gf     = (float*)(ws + off);  off += SZ_G;
    float*  Gb     = (float*)(ws + off);  off += SZ_G;
    __bf16* flatb  = (__bf16*)(ws + off); off += SZ_FLAT;
    float*  y      = (float*)(ws + off);  // BATCH*HIDDEN*4

    // 1. convert to bf16
    auto cvt = [&](const float* src, __bf16* dst, long n) {
        long blocks = (n + 255) / 256;
        f2bf_kernel<<<dim3((unsigned)blocks), 256, 0, stream>>>(src, dst, n);
    };
    cvt(x,     xb,     (long)BATCH * HIDDEN);
    cvt(Wih_f, WihFb,  (long)G4 * CHUNK);
    cvt(Wih_b, WihBb,  (long)G4 * CHUNK);
    cvt(Whh_f, WhhFb,  (long)G4 * LH);
    cvt(Whh_b, WhhBb,  (long)G4 * LH);
    cvt(projW, projWb, (long)HIDDEN * TOTAL);

    addvec_kernel<<<dim3(G4 / 256), 256, 0, stream>>>(bih_f, bhh_f, bsumF, G4);
    addvec_kernel<<<dim3(G4 / 256), 256, 0, stream>>>(bih_b, bhh_b, bsumB, G4);

    // 2. input-gate GEMMs: G[dir][t] = x_chunk_t @ Wih_dir^T + (bih+bhh)
    for (int dir = 0; dir < 2; ++dir) {
        const __bf16* W = dir ? WihBb : WihFb;
        const float* bs = dir ? bsumB : bsumF;
        float* Gp       = dir ? Gb    : Gf;
        for (int t = 0; t < CHUNKS; ++t) {
            gemm_bt_bf16<<<dim3(BATCH / 32, G4 / 512), 256, 0, stream>>>(
                xb + (size_t)t * CHUNK, HIDDEN,
                W, CHUNK, bs,
                Gp + (size_t)t * BATCH * G4, G4, CHUNK);
        }
    }

    // 3. recurrence (both directions), writes flat bf16 [B, 2048]
    constexpr size_t SMEM = 32 * 256 * 2 + 32 * 256 * 4 + 32 * (size_t)G4 * 4; // 180224
    lstm_recurrence<<<dim3(BATCH / 32), 1024, SMEM, stream>>>(
        WhhFb, WhhBb, Gf, Gb, flatb);

    // 4. projection GEMM: y = flat @ projW^T + projb
    gemm_bt_bf16<<<dim3(BATCH / 32, HIDDEN / 512), 256, 0, stream>>>(
        flatb, TOTAL, projWb, TOTAL, projb, y, HIDDEN, TOTAL);

    // 5. LayerNorm + GELU + residual
    ln_gelu_residual<<<dim3(BATCH), 256, 0, stream>>>(y, x, ln_g, ln_b, out);
}
